// Model_40656160424195
// MI455X (gfx1250) — compile-verified, matrix operand order checked by reading, not run on hardware
//
#include <hip/hip_runtime.h>

// ---------------------------------------------------------------------------
// CDNA5 (gfx1250) GNN forward. Edge MLP + post MLP run on v_wmma_f32_16x16x32_f16
// (f32 accumulate). Compute-bound workload (~43 GFLOP, ~7MB inputs) -> WMMA path.
// ---------------------------------------------------------------------------

typedef __attribute__((ext_vector_type(16))) _Float16 v16h;
typedef __attribute__((ext_vector_type(8)))  float    v8f;

#define BATCH 256
#define LHIST 11
#define NAG   64
#define MSGP  64          // msg padded 33 -> 64
#define R2C   4.0f        // COMM_RANGE^2

// ------------------------- fragment helpers --------------------------------
// A/B fragment (16-bit, 16x32 tile) per ISA 7.12.2:
//   lane<16 : row = lane,     K = k0+0..7  and k0+16..23
//   lane>=16: row = lane-16,  K = k0+8..15 and k0+24..31
__device__ __forceinline__ v16h frag_ld(const _Float16* s, int row0, int k0, int ld) {
  int lane = threadIdx.x & 31;
  int r  = row0 + (lane & 15);
  int kk = k0 + ((lane & 16) ? 8 : 0);
  const _Float16* p = s + r * ld + kk;
  v16h f;
#pragma unroll
  for (int e = 0; e < 8; e++) f[e] = p[e];
#pragma unroll
  for (int e = 0; e < 8; e++) f[8 + e] = p[16 + e];
  return f;
}

// C/D fragment (f32 16x16): lane gives N = lane&15; VGPR r gives M = r + 8*(lane>=16)
__device__ __forceinline__ void frag_st_f16(_Float16* s, v8f d, int row0, int col0,
                                            int ld, bool relu) {
  int lane = threadIdx.x & 31;
  int col = col0 + (lane & 15);
  int r0  = row0 + ((lane & 16) ? 8 : 0);
#pragma unroll
  for (int r = 0; r < 8; r++) {
    float v = d[r];
    if (relu && v < 0.f) v = 0.f;
    s[(r0 + r) * ld + col] = (_Float16)v;
  }
}

__device__ __forceinline__ void frag_st_f32(float* s, v8f d, int row0, int col0,
                                            int ld, bool relu) {
  int lane = threadIdx.x & 31;
  int col = col0 + (lane & 15);
  int r0  = row0 + ((lane & 16) ? 8 : 0);
#pragma unroll
  for (int r = 0; r < 8; r++) {
    float v = d[r];
    if (relu && v < 0.f) v = 0.f;
    s[(r0 + r) * ld + col] = v;
  }
}

// One dense layer: rows [mrow, mrow+16) of a 64-wide activation tile through a
// 64x64 weight (wT = f16 [out][K] row-major). Bias in f32; ReLU fused on store.
__device__ __forceinline__ void wmma_layer_f16(const _Float16* actIn, const _Float16* wT,
                                               const float* bias, _Float16* actOut,
                                               int mrow, bool relu) {
#pragma unroll
  for (int n = 0; n < 4; n++) {
    float bv = bias[n * 16 + (threadIdx.x & 15)];
    v8f c;
#pragma unroll
    for (int r = 0; r < 8; r++) c[r] = bv;
#pragma unroll
    for (int ks = 0; ks < 64; ks += 32) {
      v16h a = frag_ld(actIn, mrow, ks, 64);
      v16h b = frag_ld(wT, n * 16, ks, 64);
      c = __builtin_amdgcn_wmma_f32_16x16x32_f16(false, a, false, b, (short)0, c,
                                                 false, false);
    }
    frag_st_f16(actOut, c, mrow, n * 16, 64, relu);
  }
}

__device__ __forceinline__ void wmma_layer_f32out(const _Float16* actIn, const _Float16* wT,
                                                  const float* bias, float* actOut,
                                                  int mrow, bool relu) {
#pragma unroll
  for (int n = 0; n < 4; n++) {
    float bv = bias[n * 16 + (threadIdx.x & 15)];
    v8f c;
#pragma unroll
    for (int r = 0; r < 8; r++) c[r] = bv;
#pragma unroll
    for (int ks = 0; ks < 64; ks += 32) {
      v16h a = frag_ld(actIn, mrow, ks, 64);
      v16h b = frag_ld(wT, n * 16, ks, 64);
      c = __builtin_amdgcn_wmma_f32_16x16x32_f16(false, a, false, b, (short)0, c,
                                                 false, false);
    }
    frag_st_f32(actOut, c, mrow, n * 16, 64, relu);
  }
}

// ------------------------- parameter structs -------------------------------
struct EncP { const float *w0,*w1,*w2,*w3,*b0,*b1,*b2,*b3; };
struct W6   { const float* w[6]; };          // [branch*3 + layer]
struct B6   { const float* b[6]; };          // [branch*3 + layer]
struct PostArgs {
  const float* b0[2]; const float* b1[2]; const float* b2[2];
  const float* w3[2]; const float* b3[2];
};

// ------------------------- kernel 0: weight prep ---------------------------
// f32 [K][64] -> f16 [64 out][64 K] row-major, zero-padded K (33 -> 64 for gnn L0)
__global__ void k_prep_weights(W6 gnn, W6 post, _Float16* wgnn, _Float16* wpost) {
  int tid = blockIdx.x * blockDim.x + threadIdx.x;
  if (tid >= 12 * 4096) return;
  int m = tid >> 12;          // 0..11
  int e = tid & 4095;
  int outr = e >> 6, k = e & 63;
  bool isPost = (m >= 6);
  int mm = isPost ? m - 6 : m;
  const float* src = isPost ? post.w[mm] : gnn.w[mm];
  int kin = (!isPost && (mm % 3) == 0) ? 33 : 64;
  float val = (k < kin) ? src[k * 64 + outr] : 0.f;
  _Float16* dst = (isPost ? wpost : wgnn) + (size_t)mm * 4096;
  dst[outr * 64 + k] = (_Float16)val;
}

// --------------------- kernel 1: gather + encoder MLP ----------------------
__global__ void k_gather_encode(const float* obs_pos, const float* obs_vel,
                                const float* obs_goal, const float* obs_time,
                                const int* indexes, EncP pi, EncP vv,
                                float* msg_ws, float* pos_ws) {
  int tid = blockIdx.x * blockDim.x + threadIdx.x;
  if (tid >= 2 * BATCH * NAG) return;
  int branch = tid >> 14;
  int bi = tid & (BATCH * NAG - 1);
  int b = bi >> 6, n = bi & 63;
  int l = indexes[b * NAG + n];
  int base = (b * LHIST + l) * NAG + n;
  float px = obs_pos[base * 2 + 0], py = obs_pos[base * 2 + 1];
  float vx = obs_vel[base * 2 + 0], vy = obs_vel[base * 2 + 1];
  float gx = obs_goal[base * 2 + 0], gy = obs_goal[base * 2 + 1];
  float t  = obs_time[base];
  float x[6] = {gx - px, gy - py, px, py, px + vx, py + vy};
  const EncP& P = branch ? vv : pi;
  float h1[16], h2[32], h3[32];
  for (int c = 0; c < 16; c++) {
    float a = P.b0[c];
    for (int r = 0; r < 6; r++) a += x[r] * P.w0[r * 16 + c];
    h1[c] = fmaxf(a, 0.f);
  }
  for (int c = 0; c < 32; c++) {
    float a = P.b1[c];
    for (int r = 0; r < 16; r++) a += h1[r] * P.w1[r * 32 + c];
    h2[c] = fmaxf(a, 0.f);
  }
  for (int c = 0; c < 32; c++) {
    float a = P.b2[c];
    for (int r = 0; r < 32; r++) a += h2[r] * P.w2[r * 32 + c];
    h3[c] = fmaxf(a, 0.f);
  }
  float* m = msg_ws + (size_t)(branch * BATCH * NAG + bi) * MSGP;
  for (int c = 0; c < 32; c++) {
    float a = P.b3[c];
    for (int r = 0; r < 32; r++) a += h3[r] * P.w3[r * 32 + c];
    m[c] = a;                      // enc (no relu on last encoder layer)
  }
  m[32] = t;                       // msg = concat(enc, t)
  for (int c = 33; c < MSGP; c++) m[c] = 0.f;
  if (branch == 0) { pos_ws[bi * 2 + 0] = px; pos_ws[bi * 2 + 1] = py; }
}

// ----------------- kernel 2: edge MLP (WMMA) + masked aggregate ------------
// One WG per (branch, b, i): 64 edge rows (j) x 64 features; 4 waves x 16 rows.
__global__ __launch_bounds__(128) void k_edges(const float* msg_ws, const float* pos_ws,
                                               const _Float16* wgnn, B6 gb,
                                               float* agg_ws) {
  __shared__ _Float16 actA[64 * 64];
  __shared__ _Float16 actB[64 * 64];
  __shared__ float    out3[64 * 64];
  __shared__ float    maskf[64];
  __shared__ float    part[128];

  int wg = blockIdx.x;
  int branch = wg >> 14;
  int bi = wg & 16383;
  int b = bi >> 6, i = bi & 63;
  int t = threadIdx.x;
  int wave = t >> 5;

  const float* msgB = msg_ws + (size_t)(branch * 16384 + b * 64) * MSGP;
  const float* mi = msgB + i * MSGP;
  // diff tile: actA[j][k] = msg_j[k] - msg_i[k]  (padding stays 0)
  for (int e = t; e < 64 * 64; e += 128) {
    int j = e >> 6, k = e & 63;
    actA[e] = (_Float16)(msgB[j * MSGP + k] - mi[k]);
  }
  if (t < 64) {
    float dx = pos_ws[(b * 64 + t) * 2 + 0] - pos_ws[(b * 64 + i) * 2 + 0];
    float dy = pos_ws[(b * 64 + t) * 2 + 1] - pos_ws[(b * 64 + i) * 2 + 1];
    maskf[t] = (dx * dx + dy * dy <= R2C) ? 1.f : 0.f;
  }
  __syncthreads();

  int mrow = wave * 16;
  const _Float16* w0 = wgnn + (size_t)(branch * 3 + 0) * 4096;
  const _Float16* w1 = wgnn + (size_t)(branch * 3 + 1) * 4096;
  const _Float16* w2 = wgnn + (size_t)(branch * 3 + 2) * 4096;
  wmma_layer_f16(actA, w0, gb.b[branch * 3 + 0], actB, mrow, true);
  __syncthreads();
  wmma_layer_f16(actB, w1, gb.b[branch * 3 + 1], actA, mrow, true);
  __syncthreads();
  wmma_layer_f32out(actA, w2, gb.b[branch * 3 + 2], out3, mrow, false);
  __syncthreads();

  // agg[f] = sum_j mask(i,j) * m[j][f]   (split j over two 64-thread halves of the WG)
  {
    int f = t & 63, h = t >> 6;
    float s = 0.f;
    for (int j = h * 32; j < h * 32 + 32; j++) s += maskf[j] * out3[j * 64 + f];
    part[t] = s;
  }
  __syncthreads();
  if (t < 64)
    agg_ws[(size_t)(branch * 16384 + bi) * 64 + t] = part[t] + part[64 + t];
}

// ------------------- kernel 3: post MLP (WMMA) + final layer ---------------
__global__ __launch_bounds__(128) void k_post(const float* agg_ws, const _Float16* wpost,
                                              PostArgs pa, float* out) {
  __shared__ _Float16 actA[64 * 64];
  __shared__ _Float16 actB[64 * 64];
  __shared__ float    out3[64 * 64];
  int wg = blockIdx.x;            // 512 WGs: branch * 256 + tile
  int branch = wg >> 8;
  int tile = wg & 255;
  int t = threadIdx.x, wave = t >> 5;
  const float* aggB = agg_ws + (size_t)(branch * 16384 + tile * 64) * 64;
  for (int e = t; e < 4096; e += 128) actA[e] = (_Float16)aggB[e];
  __syncthreads();

  int mrow = wave * 16;
  const _Float16* w0 = wpost + (size_t)(branch * 3 + 0) * 4096;
  const _Float16* w1 = wpost + (size_t)(branch * 3 + 1) * 4096;
  const _Float16* w2 = wpost + (size_t)(branch * 3 + 2) * 4096;
  wmma_layer_f16(actA, w0, pa.b0[branch], actB, mrow, true);
  __syncthreads();
  wmma_layer_f16(actB, w1, pa.b1[branch], actA, mrow, true);
  __syncthreads();
  wmma_layer_f32out(actA, w2, pa.b2[branch], out3, mrow, true);  // relu (layer idx 2 of 4)
  __syncthreads();

  // final 64 -> {2,1} layer, f32
  int row = t & 63;
  int o = t >> 6;
  int nout = branch ? 1 : 2;
  if (o < nout) {
    const float* w3 = pa.w3[branch];
    float a = pa.b3[branch][o];
    for (int k = 0; k < 64; k++) a += out3[row * 64 + k] * w3[k * nout + o];
    int gr = tile * 64 + row;                   // = b*64 + i
    if (branch == 0) {
      int bb = gr >> 6, ii = gr & 63;
      out[bb * 128 + ii * 2 + o] = a;           // outputs [B, N*2]
    } else {
      out[32768 + gr] = a;                      // values  [B, N]
    }
  }
}

// ------------------------------- launch ------------------------------------
extern "C" void kernel_launch(void* const* d_in, const int* in_sizes, int n_in,
                              void* d_out, int out_size, void* d_ws, size_t ws_size,
                              hipStream_t stream) {
  (void)in_sizes; (void)n_in; (void)out_size; (void)ws_size;
  const float* obs_pos  = (const float*)d_in[0];
  const float* obs_vel  = (const float*)d_in[1];
  const float* obs_goal = (const float*)d_in[2];
  const float* obs_time = (const float*)d_in[3];
  const int*   indexes  = (const int*)d_in[4];

  // params_{pi,v} pytree (dict keys alphabetical): enc_b[4], enc_w[4],
  // gnn_b[3], gnn_w[3], post_b[4], post_w[4]  -> 22 arrays each
  const int P = 5, V = 27;
  auto F = [&](int i) { return (const float*)d_in[i]; };

  EncP encPi = {F(P+4), F(P+5), F(P+6), F(P+7), F(P+0), F(P+1), F(P+2), F(P+3)};
  EncP encV  = {F(V+4), F(V+5), F(V+6), F(V+7), F(V+0), F(V+1), F(V+2), F(V+3)};

  W6 gnnW = {{F(P+11), F(P+12), F(P+13), F(V+11), F(V+12), F(V+13)}};
  B6 gnnB = {{F(P+8),  F(P+9),  F(P+10), F(V+8),  F(V+9),  F(V+10)}};
  W6 postW = {{F(P+18), F(P+19), F(P+20), F(V+18), F(V+19), F(V+20)}};
  PostArgs pa;
  pa.b0[0] = F(P+14); pa.b0[1] = F(V+14);
  pa.b1[0] = F(P+15); pa.b1[1] = F(V+15);
  pa.b2[0] = F(P+16); pa.b2[1] = F(V+16);
  pa.w3[0] = F(P+21); pa.w3[1] = F(V+21);
  pa.b3[0] = F(P+17); pa.b3[1] = F(V+17);

  // workspace layout
  char* ws = (char*)d_ws;
  float*     msg_ws = (float*)ws;                                   // 8 MB
  float*     agg_ws = (float*)(ws + 8388608);                       // 8 MB
  float*     pos_ws = (float*)(ws + 16777216);                      // 128 KB
  _Float16*  wgnn   = (_Float16*)(ws + 16908288);                   // 48 KB
  _Float16*  wpost  = (_Float16*)(ws + 16957440);                   // 48 KB

  k_prep_weights<<<(12 * 4096 + 255) / 256, 256, 0, stream>>>(gnnW, postW, wgnn, wpost);
  k_gather_encode<<<(2 * BATCH * NAG + 255) / 256, 256, 0, stream>>>(
      obs_pos, obs_vel, obs_goal, obs_time, indexes, encPi, encV, msg_ws, pos_ws);
  k_edges<<<2 * BATCH * NAG, 128, 0, stream>>>(msg_ws, pos_ws, wgnn, gnnB, agg_ws);
  k_post<<<512, 128, 0, stream>>>(agg_ws, wpost, pa, (float*)d_out);
}
